// Informer_89867895701559
// MI455X (gfx1250) — matrix operand order repaired and checked by reading, not compiled, not run on hardware
//
#include <hip/hip_runtime.h>
#include <hip/hip_bf16.h>
#include <math.h>

// ---------------- model constants ----------------
constexpr int BN   = 512;   // batch
constexpr int SEQ  = 70;
constexpr int CIN  = 20;
constexpr int DM   = 512;   // model dim
constexpr int HH   = 8;     // heads
constexpr int HE   = 64;    // head dim
constexpr int DFFV = 512;
constexpr float EPSV = 1e-5f;
constexpr int L0 = 70, L1 = 35, L2 = 18;
constexpr int U0 = 25, U1 = 20, U2 = 15;
constexpr int DECL = 18, UD = 15;
constexpr int KCONV = 3 * DM;   // 1536, im2col K for distill convs

typedef __attribute__((ext_vector_type(16))) _Float16 v16h;
typedef __attribute__((ext_vector_type(8)))  float    v8f;

// =====================================================================
// Register-blocked WMMA GEMM:  C[M,N] = act( A[M,K] @ W[N,K]^T + bias[N] )
// One wave computes a 64x32 C macro-tile (4x2 tiles of 16x16).
// Requires: M % 64 == 0, N % 32 == 0, K % 32 == 0 (true for every call here).
// act: 0 = none, 1 = exact gelu
// =====================================================================
constexpr int WMT = 4;   // 16-row tiles per wave (M)
constexpr int WNT = 2;   // 16-col tiles per wave (N)

__device__ __forceinline__ v16h load_afrag(const float* __restrict__ arow, int k0, int hs) {
  // A 16x32 f16 layout: lanes0-15 K=2j,2j+1 (j<4) then 16+2(j-4); lanes16-31 +8
  v16h a;
  const int abase = k0 + hs * 8;
#pragma unroll
  for (int j = 0; j < 4; ++j) {
    int kk = abase + 2 * j;
    a[2*j]   = (_Float16)arow[kk];
    a[2*j+1] = (_Float16)arow[kk+1];
  }
#pragma unroll
  for (int j = 4; j < 8; ++j) {
    int kk = abase + 16 + 2 * (j - 4);
    a[2*j]   = (_Float16)arow[kk];
    a[2*j+1] = (_Float16)arow[kk+1];
  }
  return a;
}

__device__ __forceinline__ v16h load_bfrag(const float* __restrict__ wrow, int k0, int hs) {
  // B 32x16 f16 layout: lanes0-15 K=2j,2j+1 (j<4) then 16+2(j-4); lanes16-31 +8
  v16h b;
  const int bbase = k0 + hs * 8;
#pragma unroll
  for (int j = 0; j < 8; ++j) {
    int kk = bbase + 2 * j + (j >= 4 ? 8 : 0);
    b[2*j]   = (_Float16)wrow[kk];
    b[2*j+1] = (_Float16)wrow[kk+1];
  }
  return b;
}

__global__ void __launch_bounds__(32)
wmma_linear(const float* __restrict__ A, const float* __restrict__ Wm,
            const float* __restrict__ bias, float* __restrict__ C,
            int M, int N, int K, int act) {
  const int lane = threadIdx.x;
  const int hs   = lane >> 4;       // half-wave select
  const int lid  = lane & 15;
  const int tm0 = blockIdx.y * WMT;
  const int tn0 = blockIdx.x * WNT;

  const float* arow[WMT];
  const float* wrow[WNT];
#pragma unroll
  for (int i = 0; i < WMT; ++i)
    arow[i] = A + ((size_t)(tm0 + i) * 16 + lid) * K;
#pragma unroll
  for (int j = 0; j < WNT; ++j)
    wrow[j] = Wm + ((size_t)(tn0 + j) * 16 + lid) * K;

  v8f acc[WMT][WNT] = {};
  for (int k0 = 0; k0 < K; k0 += 32) {
    v16h af[WMT], bf[WNT];
#pragma unroll
    for (int i = 0; i < WMT; ++i) af[i] = load_afrag(arow[i], k0, hs);
#pragma unroll
    for (int j = 0; j < WNT; ++j) bf[j] = load_bfrag(wrow[j], k0, hs);
    if (k0 + 32 < K) {                    // hint next K tile toward the caches
      __builtin_prefetch(arow[0] + k0 + 32, 0, 1);
      __builtin_prefetch(wrow[0] + k0 + 32, 0, 1);
    }
#pragma unroll
    for (int i = 0; i < WMT; ++i)
#pragma unroll
      for (int j = 0; j < WNT; ++j)
        acc[i][j] = __builtin_amdgcn_wmma_f32_16x16x32_f16(
            false, af[i], false, bf[j], (short)0, acc[i][j], false, false);
  }

  // C 16x16 f32 layout: VGPR r -> row r (lanes0-15) / row r+8 (lanes16-31)
#pragma unroll
  for (int i = 0; i < WMT; ++i) {
#pragma unroll
    for (int r = 0; r < 8; ++r) {
      int row = (tm0 + i) * 16 + r + 8 * hs;
#pragma unroll
      for (int j = 0; j < WNT; ++j) {
        int col = (tn0 + j) * 16 + lid;
        float v = acc[i][j][r] + bias[col];
        if (act == 1) v = 0.5f * v * (1.0f + erff(v * 0.70710678118654752f));
        C[(size_t)row * N + col] = v;
      }
    }
  }
}

// ---------------- token embedding: circular conv(20->512,k=3) + sinusoid pos ----
__global__ void token_embed(const float* __restrict__ X, const float* __restrict__ Wt,
                            float* __restrict__ Out) {
  size_t t = (size_t)blockIdx.x * blockDim.x + threadIdx.x;
  size_t total = (size_t)BN * SEQ * DM;
  if (t >= total) return;
  int cch = (int)(t % DM);
  int l   = (int)((t / DM) % SEQ);
  int b   = (int)(t / ((size_t)DM * SEQ));
  float s = 0.f;
  for (int kt = 0; kt < 3; ++kt) {
    int tl = l + kt - 1;
    tl = (tl < 0) ? SEQ - 1 : (tl >= SEQ ? 0 : tl);
    const float* xp = X + ((size_t)b * SEQ + tl) * CIN;
    const float* wp = Wt + ((size_t)cch * CIN) * 3 + kt;
    for (int ci = 0; ci < CIN; ++ci) s += xp[ci] * wp[ci * 3];
  }
  int j = cch >> 1;
  float div = expf(-(float)(2 * j) * (logf(10000.f) / (float)DM));
  float ang = (float)l * div;
  s += (cch & 1) ? cosf(ang) : sinf(ang);
  Out[t] = s;
}

// ---------------- QK^T scores: S[b,h,i,j] over head dim 64 -----------------
__global__ void qk_scores(const float* __restrict__ Q, const float* __restrict__ Km,
                          float* __restrict__ S, int L) {
  size_t idx = (size_t)blockIdx.x * blockDim.x + threadIdx.x;
  size_t total = (size_t)BN * HH * L * L;
  if (idx >= total) return;
  int j = (int)(idx % L);
  size_t t = idx / L;
  int i = (int)(t % L); t /= L;
  int h = (int)(t % HH);
  int b = (int)(t / HH);
  const float* q = Q  + ((size_t)(b * L + i)) * DM + h * HE;
  const float* k = Km + ((size_t)(b * L + j)) * DM + h * HE;
  float s = 0.f;
  for (int e = 0; e < HE; ++e) s += q[e] * k[e];
  S[idx] = s;
}

// ---------------- ProbSparse metric M = max(sampled) - sum(sampled)/L, then top-U
__global__ void metric_topk(const float* __restrict__ S, const int* __restrict__ samp,
                            int* __restrict__ idxOut, int L, int U) {
  int t = blockIdx.x * blockDim.x + threadIdx.x;
  if (t >= BN * HH) return;
  const float* sc = S + (size_t)t * L * L;
  float Mloc[70];
  bool used[70];
  for (int l = 0; l < L; ++l) {
    const float* row = sc + (size_t)l * L;
    float mx = -INFINITY, sm = 0.f;
    for (int s = 0; s < U; ++s) { float v = row[samp[l * U + s]]; mx = fmaxf(mx, v); sm += v; }
    Mloc[l] = mx - sm / (float)L;
    used[l] = false;
  }
  for (int u = 0; u < U; ++u) {
    int best = 0; float bv = -INFINITY;
    for (int l = 0; l < L; ++l)
      if (!used[l] && Mloc[l] > bv) { bv = Mloc[l]; best = l; }
    used[best] = true;
    idxOut[(size_t)t * U + u] = best;
  }
}

// ---------------- context init: mean-of-V broadcast  (non-causal) --------------
__global__ void ctx_mean(const float* __restrict__ V, float* __restrict__ ctx, int L) {
  int t = blockIdx.x * blockDim.x + threadIdx.x;
  if (t >= BN * HH * HE) return;
  int e = t % HE, h = (t / HE) % HH, b = t / (HE * HH);
  float s = 0.f;
  for (int l = 0; l < L; ++l) s += V[((size_t)(b * L + l)) * DM + h * HE + e];
  s /= (float)L;
  float* c = ctx + ((size_t)(b * HH + h) * L) * HE + e;
  for (int l = 0; l < L; ++l) c[(size_t)l * HE] = s;
}

// ---------------- context init: cumsum-of-V (causal) -------------------------
__global__ void ctx_cumsum(const float* __restrict__ V, float* __restrict__ ctx, int L) {
  int t = blockIdx.x * blockDim.x + threadIdx.x;
  if (t >= BN * HH * HE) return;
  int e = t % HE, h = (t / HE) % HH, b = t / (HE * HH);
  float s = 0.f;
  float* c = ctx + ((size_t)(b * HH + h) * L) * HE + e;
  for (int l = 0; l < L; ++l) {
    s += V[((size_t)(b * L + l)) * DM + h * HE + e];
    c[(size_t)l * HE] = s;
  }
}

// ---------------- softmax over selected rows + V-update + scatter into ctx ----
__global__ void __launch_bounds__(128)
prob_attn_update(const float* __restrict__ S, const float* __restrict__ V,
                 const int* __restrict__ idx, float* __restrict__ ctx,
                 int L, int U, int causal) {
  int t = blockIdx.x;
  int u = t % U; int h = (t / U) % HH; int b = t / (U * HH);
  __shared__ float p[72];
  __shared__ float red[128];
  int i = idx[((size_t)(b * HH + h)) * U + u];
  const float* row = S + (((size_t)(b * HH + h) * L) + i) * L;
  const float scale = 0.125f;       // 1/sqrt(64)
  int tid = threadIdx.x;
  for (int j = tid; j < L; j += blockDim.x) {
    float v = row[j] * scale;
    if (causal && j > i) v = -INFINITY;
    p[j] = v;
  }
  __syncthreads();
  float mx = -INFINITY;
  for (int j = tid; j < L; j += blockDim.x) mx = fmaxf(mx, p[j]);
  red[tid] = mx; __syncthreads();
  for (int s = 64; s > 0; s >>= 1) { if (tid < s) red[tid] = fmaxf(red[tid], red[tid + s]); __syncthreads(); }
  mx = red[0]; __syncthreads();
  float sm = 0.f;
  for (int j = tid; j < L; j += blockDim.x) { float e = expf(p[j] - mx); p[j] = e; sm += e; }
  red[tid] = sm; __syncthreads();
  for (int s = 64; s > 0; s >>= 1) { if (tid < s) red[tid] += red[tid + s]; __syncthreads(); }
  sm = red[0]; __syncthreads();
  if (tid < HE) {
    float acc = 0.f;
    for (int j = 0; j < L; ++j) acc += p[j] * V[((size_t)(b * L + j)) * DM + h * HE + tid];
    ctx[(((size_t)(b * HH + h) * L) + i) * HE + tid] = acc / sm;
  }
}

// ---------------- ctx (B,H,L,E) -> rows (B*L, H*E)  (mix=False path) ----------
__global__ void ctx_to_rows(const float* __restrict__ ctx, float* __restrict__ outm, int L) {
  size_t t = (size_t)blockIdx.x * blockDim.x + threadIdx.x;
  size_t total = (size_t)BN * L * DM;
  if (t >= total) return;
  int cch = (int)(t % DM);
  int l   = (int)((t / DM) % L);
  int b   = (int)(t / ((size_t)DM * L));
  int h = cch / HE, e = cch % HE;
  outm[t] = ctx[(((size_t)(b * HH + h) * L) + l) * HE + e];
}

// ---------------- LayerNorm over last dim 512, optional residual --------------
__global__ void __launch_bounds__(128)
layernorm_k(const float* __restrict__ X, const float* __restrict__ Res,
            const float* __restrict__ g, const float* __restrict__ be,
            float* __restrict__ Out) {
  int row = blockIdx.x, tid = threadIdx.x;
  __shared__ float red[128];
  const float* x = X + (size_t)row * DM;
  const float* r = Res ? Res + (size_t)row * DM : nullptr;
  float vals[4]; float s = 0.f;
#pragma unroll
  for (int j = 0; j < 4; ++j) {
    int cc = tid * 4 + j;
    float v = x[cc] + (r ? r[cc] : 0.f);
    vals[j] = v; s += v;
  }
  red[tid] = s; __syncthreads();
  for (int st = 64; st > 0; st >>= 1) { if (tid < st) red[tid] += red[tid + st]; __syncthreads(); }
  float mean = red[0] / (float)DM; __syncthreads();
  s = 0.f;
#pragma unroll
  for (int j = 0; j < 4; ++j) { float d = vals[j] - mean; s += d * d; }
  red[tid] = s; __syncthreads();
  for (int st = 64; st > 0; st >>= 1) { if (tid < st) red[tid] += red[tid + st]; __syncthreads(); }
  float inv = rsqrtf(red[0] / (float)DM + EPSV); __syncthreads();
#pragma unroll
  for (int j = 0; j < 4; ++j) {
    int cc = tid * 4 + j;
    Out[(size_t)row * DM + cc] = (vals[j] - mean) * inv * g[cc] + be[cc];
  }
}

// ---------------- distill conv as GEMM: im2col (circular, k=3) ----------------
__global__ void im2col3_circ(const float* __restrict__ X, float* __restrict__ A2, int L) {
  // X (B,L,512) -> A2 (B*L, 1536): A2[(b*L+l)*1536 + kt*512 + ci] = X[b, wrap(l+kt-1), ci]
  size_t t = (size_t)blockIdx.x * blockDim.x + threadIdx.x;
  size_t total = (size_t)BN * L * KCONV;
  if (t >= total) return;
  int c  = (int)(t % KCONV);
  int l  = (int)((t / KCONV) % L);
  int b  = (int)(t / ((size_t)KCONV * L));
  int kt = c / DM, ci = c % DM;
  int tl = l + kt - 1;
  tl = (tl < 0) ? L - 1 : (tl >= L ? 0 : tl);
  A2[t] = X[((size_t)(b * L + tl)) * DM + ci];
}

__global__ void conv_w_repack(const float* __restrict__ Wc, float* __restrict__ Wr) {
  // Wc (512,512,3) -> Wr (512, 1536): Wr[co][kt*512+ci] = Wc[co][ci][kt]
  size_t t = (size_t)blockIdx.x * blockDim.x + threadIdx.x;
  size_t total = (size_t)DM * KCONV;
  if (t >= total) return;
  int c  = (int)(t % KCONV);
  int co = (int)(t / KCONV);
  int kt = c / DM, ci = c % DM;
  Wr[t] = Wc[((size_t)co * DM + ci) * 3 + kt];
}

// ---------------- BN over (B,L) per channel, channel-last (B,L,512) layout ----
__global__ void __launch_bounds__(256)
bn_stats_cl(const float* __restrict__ Y, float* __restrict__ mv, float* __restrict__ vv, int L) {
  int co = blockIdx.x, tid = threadIdx.x;
  __shared__ float r1[256], r2[256];
  float s = 0.f, s2 = 0.f;
  int Nl = BN * L;
  for (int t = tid; t < Nl; t += 256) {
    float v = Y[(size_t)t * DM + co];
    s += v; s2 += v * v;
  }
  r1[tid] = s; r2[tid] = s2; __syncthreads();
  for (int st = 128; st > 0; st >>= 1) {
    if (tid < st) { r1[tid] += r1[tid + st]; r2[tid] += r2[tid + st]; }
    __syncthreads();
  }
  if (tid == 0) { float m = r1[0] / Nl; mv[co] = m; vv[co] = r2[0] / Nl - m * m; }
}

__global__ void bn_elu_cl(float* __restrict__ Y, const float* __restrict__ mv,
                          const float* __restrict__ vv, const float* __restrict__ g,
                          const float* __restrict__ be, int L) {
  size_t t = (size_t)blockIdx.x * blockDim.x + threadIdx.x;
  size_t total = (size_t)BN * L * DM;
  if (t >= total) return;
  int co = (int)(t % DM);
  float v = (Y[t] - mv[co]) * rsqrtf(vv[co] + EPSV) * g[co] + be[co];
  Y[t] = v > 0.f ? v : (expf(v) - 1.f);
}

// ---------------- maxpool k=3 s=2 pad=1 over time, channel-last ---------------
__global__ void maxpool_cl(const float* __restrict__ Y, float* __restrict__ Out, int L, int Lout) {
  size_t t = (size_t)blockIdx.x * blockDim.x + threadIdx.x;
  size_t total = (size_t)BN * Lout * DM;
  if (t >= total) return;
  int co = (int)(t % DM);
  int t2 = (int)((t / DM) % Lout);
  int b  = (int)(t / ((size_t)DM * Lout));
  float m = -INFINITY;
  for (int dt = -1; dt <= 1; ++dt) {
    int tt = 2 * t2 + dt;
    if (tt >= 0 && tt < L) m = fmaxf(m, Y[((size_t)(b * L + tt)) * DM + co]);
  }
  Out[t] = m;
}

// ---------------- fused full cross-attention (L=S=18) -------------------------
__global__ void __launch_bounds__(64)
full_attn(const float* __restrict__ Q, const float* __restrict__ Km,
          const float* __restrict__ V, float* __restrict__ Outm, int L, int S) {
  int t = blockIdx.x;
  int i = t % L; int h = (t / L) % HH; int b = t / (L * HH);
  __shared__ float p[32];
  int tid = threadIdx.x;
  const float* q = Q + ((size_t)(b * L + i)) * DM + h * HE;
  if (tid < S) {
    const float* k = Km + ((size_t)(b * S + tid)) * DM + h * HE;
    float s = 0.f;
    for (int e = 0; e < HE; ++e) s += q[e] * k[e];
    p[tid] = s * 0.125f;
  }
  __syncthreads();
  if (tid == 0) {
    float mx = -INFINITY;
    for (int j = 0; j < S; ++j) mx = fmaxf(mx, p[j]);
    float sm = 0.f;
    for (int j = 0; j < S; ++j) { p[j] = expf(p[j] - mx); sm += p[j]; }
    for (int j = 0; j < S; ++j) p[j] /= sm;
  }
  __syncthreads();
  float acc = 0.f;
  for (int j = 0; j < S; ++j) acc += p[j] * V[((size_t)(b * S + j)) * DM + h * HE + tid];
  Outm[((size_t)(b * L + i)) * DM + h * HE + tid] = acc;
}

// ---------------- head BN over (B, C) per position l --------------------------
__global__ void __launch_bounds__(256)
bnseq_stats(const float* __restrict__ Y, float* __restrict__ mv, float* __restrict__ vv, int C) {
  int l = blockIdx.x, tid = threadIdx.x;
  __shared__ float r1[256], r2[256];
  float s = 0.f, s2 = 0.f;
  int Nl = BN * C;
  for (int t = tid; t < Nl; t += 256) {
    int b = t / C, cc = t % C;
    float v = Y[((size_t)(b * DECL + l)) * C + cc];
    s += v; s2 += v * v;
  }
  r1[tid] = s; r2[tid] = s2; __syncthreads();
  for (int st = 128; st > 0; st >>= 1) {
    if (tid < st) { r1[tid] += r1[tid + st]; r2[tid] += r2[tid + st]; }
    __syncthreads();
  }
  if (tid == 0) { float m = r1[0] / Nl; mv[l] = m; vv[l] = r2[0] / Nl - m * m; }
}

__global__ void bnseq_relu(float* __restrict__ Y, const float* __restrict__ mv,
                           const float* __restrict__ vv, const float* __restrict__ g,
                           const float* __restrict__ be, int C) {
  size_t t = (size_t)blockIdx.x * blockDim.x + threadIdx.x;
  size_t total = (size_t)BN * DECL * C;
  if (t >= total) return;
  int l = (int)((t / C) % DECL);
  float v = (Y[t] - mv[l]) * rsqrtf(vv[l] + EPSV) * g[l] + be[l];
  Y[t] = fmaxf(v, 0.f);
}

// ---------------- small GEMV (N=1) -------------------------------------------
__global__ void gemv1(const float* __restrict__ A, const float* __restrict__ w,
                      const float* __restrict__ b, float* __restrict__ y, int M, int K) {
  int m = blockIdx.x * blockDim.x + threadIdx.x;
  if (m >= M) return;
  float s = b[0];
  for (int k = 0; k < K; ++k) s += A[(size_t)m * K + k] * w[k];
  y[m] = s;
}

// ---------------- log_softmax over positions (axis 1, L=18) -------------------
__global__ void logsoftmax_pos(const float* __restrict__ X, float* __restrict__ Out) {
  int b = blockIdx.x * blockDim.x + threadIdx.x;
  if (b >= BN) return;
  const float* x = X + (size_t)b * DECL;
  float mx = -INFINITY;
  for (int l = 0; l < DECL; ++l) mx = fmaxf(mx, x[l]);
  float sm = 0.f;
  for (int l = 0; l < DECL; ++l) sm += expf(x[l] - mx);
  float ls = logf(sm) + mx;
  for (int l = 0; l < DECL; ++l) Out[(size_t)b * DECL + l] = x[l] - ls;
}

// ---------------- deterministic sample-index tables ---------------------------
__global__ void gen_samp(int* __restrict__ out, int L, int U, unsigned seed) {
  int t = blockIdx.x * blockDim.x + threadIdx.x;
  if (t >= L * U) return;
  unsigned x = seed * 2654435761u + (unsigned)t * 2246822519u + 3266489917u;
  x ^= x >> 15; x *= 2654435761u; x ^= x >> 13; x *= 3266489917u; x ^= x >> 16;
  out[t] = (int)(x % (unsigned)L);
}

// ---------------- misc --------------------------------------------------------
__global__ void copyf(const float* __restrict__ a, float* __restrict__ b, size_t n) {
  size_t t = (size_t)blockIdx.x * blockDim.x + threadIdx.x;
  if (t < n) b[t] = a[t];
}
__global__ void take_last(const float* __restrict__ y, float* __restrict__ out) {
  int b = blockIdx.x * blockDim.x + threadIdx.x;
  if (b < BN) out[b] = y[(size_t)b * DECL + (DECL - 1)];
}

// =====================================================================
// Host orchestration
// =====================================================================
struct AttnP { const float *wq,*bq,*wk,*bk,*wv,*bv,*wo,*bo; };
struct EncP  { AttnP attn; const float *w1,*b1,*w2,*b2,*ln1g,*ln1b,*ln2g,*ln2b; };
struct ConvP { const float *w,*b,*g,*beta; };
struct DecP  { AttnP selfa, cross; const float *w1,*b1,*w2,*b2,*ln1g,*ln1b,*ln2g,*ln2b,*ln3g,*ln3b; };
struct HeadP { const float *w1,*b1,*g1,*be1,*w2,*b2,*g2,*be2,*w3,*b3; };

extern "C" void kernel_launch(void* const* d_in, const int* in_sizes, int n_in,
                              void* d_out, int out_size, void* d_ws, size_t ws_size,
                              hipStream_t stream) {
  (void)in_sizes; (void)n_in; (void)out_size; (void)ws_size;
  const float* x_in = (const float*)d_in[0];
  // d_in[1]=flag, d_in[2]=alpha (unused by forward)

  int cur = 3;
  auto nextp = [&]() { return (const float*)d_in[cur++]; };
  auto readAttn = [&](AttnP& a) {
    a.wq = nextp(); a.bq = nextp(); a.wk = nextp(); a.bk = nextp();
    a.wv = nextp(); a.bv = nextp(); a.wo = nextp(); a.bo = nextp();
  };
  auto readEnc = [&](EncP& e) {
    readAttn(e.attn);
    e.w1 = nextp(); e.b1 = nextp(); e.w2 = nextp(); e.b2 = nextp();
    e.ln1g = nextp(); e.ln1b = nextp(); e.ln2g = nextp(); e.ln2b = nextp();
  };
  auto readConv = [&](ConvP& c) { c.w = nextp(); c.b = nextp(); c.g = nextp(); c.beta = nextp(); };
  auto readDec = [&](DecP& dd) {
    readAttn(dd.selfa); readAttn(dd.cross);
    dd.w1 = nextp(); dd.b1 = nextp(); dd.w2 = nextp(); dd.b2 = nextp();
    dd.ln1g = nextp(); dd.ln1b = nextp(); dd.ln2g = nextp(); dd.ln2b = nextp();
    dd.ln3g = nextp(); dd.ln3b = nextp();
  };
  auto readHead = [&](HeadP& h) {
    h.w1 = nextp(); h.b1 = nextp(); h.g1 = nextp(); h.be1 = nextp();
    h.w2 = nextp(); h.b2 = nextp(); h.g2 = nextp(); h.be2 = nextp();
    h.w3 = nextp(); h.b3 = nextp();
  };

  const float* tok_w = nextp();
  EncP  encPriv[3];  for (int i = 0; i < 3; ++i) readEnc(encPriv[i]);
  ConvP convPriv[2]; for (int i = 0; i < 2; ++i) readConv(convPriv[i]);
  const float* png = nextp(); const float* pnb = nextp();
  EncP  encShare[3]; for (int i = 0; i < 3; ++i) readEnc(encShare[i]);
  ConvP convShare[2];for (int i = 0; i < 2; ++i) readConv(convShare[i]);
  const float* sng = nextp(); const float* snb = nextp();
  DecP  dec[2];      for (int i = 0; i < 2; ++i) readDec(dec[i]);
  const float* dng = nextp(); const float* dnb = nextp();
  const float* p1w = nextp(); const float* p1b = nextp();
  const float* p2w = nextp(); const float* p2b = nextp();
  const float* p3w = nextp(); const float* p3b = nextp();
  HeadP clsH; readHead(clsH);
  HeadP domH; readHead(domH);

  // ---------------- workspace carve-up (floats) -------------------------------
  float* Wsp = (float*)d_ws;
  size_t off = 0;
  auto alloc = [&](size_t n) { float* p = Wsp + off; off += n; return p; };
  const size_t BIG = (size_t)BN * L0 * DM;              // 18,350,080
  float* emb  = alloc(BIG);
  float* xb   = alloc(BIG);
  float* qb   = alloc(BIG);   // qb..vb contiguous: doubles as (B*L, 1536) im2col buffer
  float* kb   = alloc(BIG);
  float* vb   = alloc(BIG);
  float* ctxb = alloc(BIG);
  float* t1   = alloc(BIG);
  float* t2   = alloc(BIG);
  float* scb  = alloc((size_t)BN * HH * L0 * L0);       // 20,070,400
  float* privb  = alloc((size_t)BN * DECL * DM);
  float* shareb = alloc((size_t)BN * DECL * DM);
  int*   sampAll = (int*)alloc(8192);
  int*   idxb    = (int*)alloc((size_t)BN * HH * U0);
  float* bnm = alloc(DM); float* bnv = alloc(DM);
  float* hb1 = alloc((size_t)BN * DECL * 128);
  float* hb2 = alloc((size_t)BN * DECL * 64);
  float* hb3 = alloc((size_t)BN * DECL);
  float* hm  = alloc(DECL); float* hv = alloc(DECL);
  (void)kb; (void)vb;

  int* spPriv[3]  = { sampAll, sampAll + L0*U0, sampAll + L0*U0 + L1*U1 };
  int* spShare[3] = { spPriv[2] + L2*U2, spPriv[2] + L2*U2 + L0*U0,
                      spPriv[2] + L2*U2 + L0*U0 + L1*U1 };
  int* spDec[2]   = { spShare[2] + L2*U2, spShare[2] + L2*U2 + DECL*UD };

  auto gblk = [](size_t n) { return (unsigned)((n + 255) / 256); };
  auto gemm = [&](const float* A, const float* Wm, const float* bias, float* C,
                  int M, int N, int K, int act) {
    dim3 grid(N / (16 * WNT), M / (16 * WMT));   // 64x32 macro-tile per wave
    wmma_linear<<<grid, 32, 0, stream>>>(A, Wm, bias, C, M, N, K, act);
  };

  // sample tables
  {
    int Ls[3] = {L0, L1, L2}, Us[3] = {U0, U1, U2};
    for (int i = 0; i < 3; ++i) {
      gen_samp<<<gblk(Ls[i]*Us[i]), 256, 0, stream>>>(spPriv[i],  Ls[i], Us[i], 11u + i);
      gen_samp<<<gblk(Ls[i]*Us[i]), 256, 0, stream>>>(spShare[i], Ls[i], Us[i], 47u + i);
    }
    gen_samp<<<gblk(DECL*UD), 256, 0, stream>>>(spDec[0], DECL, UD, 101u);
    gen_samp<<<gblk(DECL*UD), 256, 0, stream>>>(spDec[1], DECL, UD, 103u);
  }

  // token conv + positional embedding
  token_embed<<<gblk(BIG), 256, 0, stream>>>(x_in, tok_w, emb);

  // ---------------- encoder ---------------------------------------------------
  auto run_encoder = [&](const EncP* layers, const ConvP* convs,
                         const float* ng, const float* nb, int** samps, float* outp) {
    copyf<<<gblk(BIG), 256, 0, stream>>>(emb, xb, BIG);
    int Ls[3] = {L0, L1, L2}, Us[3] = {U0, U1, U2};
    for (int i = 0; i < 3; ++i) {
      int L = Ls[i], U = Us[i], M = BN * L;
      const EncP& e = layers[i];
      gemm(xb, e.attn.wq, e.attn.bq, qb, M, DM, DM, 0);
      gemm(xb, e.attn.wk, e.attn.bk, kb, M, DM, DM, 0);
      gemm(xb, e.attn.wv, e.attn.bv, vb, M, DM, DM, 0);
      qk_scores<<<gblk((size_t)BN*HH*L*L), 256, 0, stream>>>(qb, kb, scb, L);
      metric_topk<<<gblk(BN*HH), 256, 0, stream>>>(scb, samps[i], idxb, L, U);
      ctx_mean<<<gblk(BN*HH*HE), 256, 0, stream>>>(vb, ctxb, L);
      prob_attn_update<<<BN*HH*U, 128, 0, stream>>>(scb, vb, idxb, ctxb, L, U, 0);
      ctx_to_rows<<<gblk((size_t)M*DM), 256, 0, stream>>>(ctxb, t1, L);
      gemm(t1, e.attn.wo, e.attn.bo, t2, M, DM, DM, 0);
      layernorm_k<<<M, 128, 0, stream>>>(t2, xb, e.ln1g, e.ln1b, t1);
      gemm(t1, e.w1, e.b1, t2, M, DFFV, DM, 1);           // gelu
      gemm(t2, e.w2, e.b2, qb, M, DM, DFFV, 0);
      layernorm_k<<<M, 128, 0, stream>>>(qb, t1, e.ln2g, e.ln2b, xb);
      if (i < 2) {
        // distill conv as WMMA GEMM over K = 3*512 (im2col), then BN+ELU+maxpool
        int Lout = (L - 1) / 2 + 1;
        im2col3_circ<<<gblk((size_t)M*KCONV), 256, 0, stream>>>(xb, qb, L);
        conv_w_repack<<<gblk((size_t)DM*KCONV), 256, 0, stream>>>(convs[i].w, t1);
        gemm(qb, t1, convs[i].b, t2, M, DM, KCONV, 0);
        bn_stats_cl<<<DM, 256, 0, stream>>>(t2, bnm, bnv, L);
        bn_elu_cl<<<gblk((size_t)M*DM), 256, 0, stream>>>(t2, bnm, bnv, convs[i].g, convs[i].beta, L);
        maxpool_cl<<<gblk((size_t)BN*Lout*DM), 256, 0, stream>>>(t2, xb, L, Lout);
      }
    }
    layernorm_k<<<BN*DECL, 128, 0, stream>>>(xb, nullptr, ng, nb, outp);
  };

  run_encoder(encPriv,  convPriv,  png, pnb, spPriv,  privb);
  run_encoder(encShare, convShare, sng, snb, spShare, shareb);

  // ---------------- decoder ---------------------------------------------------
  const size_t DEC_N = (size_t)BN * DECL * DM;
  copyf<<<gblk(DEC_N), 256, 0, stream>>>(privb, xb, DEC_N);
  for (int li = 0; li < 2; ++li) {
    const DecP& dd = dec[li];
    const int M = BN * DECL;
    // masked ProbSparse self-attention (mix=True)
    gemm(xb, dd.selfa.wq, dd.selfa.bq, qb, M, DM, DM, 0);
    gemm(xb, dd.selfa.wk, dd.selfa.bk, kb, M, DM, DM, 0);
    gemm(xb, dd.selfa.wv, dd.selfa.bv, vb, M, DM, DM, 0);
    qk_scores<<<gblk((size_t)BN*HH*DECL*DECL), 256, 0, stream>>>(qb, kb, scb, DECL);
    metric_topk<<<gblk(BN*HH), 256, 0, stream>>>(scb, spDec[li], idxb, DECL, UD);
    ctx_cumsum<<<gblk(BN*HH*HE), 256, 0, stream>>>(vb, ctxb, DECL);
    prob_attn_update<<<BN*HH*UD, 128, 0, stream>>>(scb, vb, idxb, ctxb, DECL, UD, 1);
    // mix=True: (B,H,L,E) buffer reinterprets directly as (B*L, 512) rows
    gemm(ctxb, dd.selfa.wo, dd.selfa.bo, t2, M, DM, DM, 0);
    layernorm_k<<<M, 128, 0, stream>>>(t2, xb, dd.ln1g, dd.ln1b, t1);
    // full cross attention vs share
    gemm(t1, dd.cross.wq, dd.cross.bq, qb, M, DM, DM, 0);
    gemm(shareb, dd.cross.wk, dd.cross.bk, kb, M, DM, DM, 0);
    gemm(shareb, dd.cross.wv, dd.cross.bv, vb, M, DM, DM, 0);
    full_attn<<<BN*HH*DECL, 64, 0, stream>>>(qb, kb, vb, ctxb, DECL, DECL);
    gemm(ctxb, dd.cross.wo, dd.cross.bo, t2, M, DM, DM, 0);
    layernorm_k<<<M, 128, 0, stream>>>(t2, t1, dd.ln2g, dd.ln2b, qb);
    // FFN
    gemm(qb, dd.w1, dd.b1, t1, M, DFFV, DM, 1);
    gemm(t1, dd.w2, dd.b2, t2, M, DM, DFFV, 0);
    layernorm_k<<<M, 128, 0, stream>>>(t2, qb, dd.ln3g, dd.ln3b, xb);
  }
  layernorm_k<<<BN*DECL, 128, 0, stream>>>(xb, nullptr, dng, dnb, t1);

  // projection chain -> y_rul
  float* out_f = (float*)d_out;
  float* seg_yrul  = out_f;
  float* seg_cls   = out_f + BN;
  float* seg_dom   = seg_cls + (size_t)BN * DECL;
  float* seg_share = seg_dom + (size_t)BN * DECL;
  float* seg_priv  = seg_share + DEC_N;

  gemm(t1, p1w, p1b, hb1, BN*DECL, 128, DM, 0);
  gemm(hb1, p2w, p2b, hb2, BN*DECL, 64, 128, 0);
  gemv1<<<gblk(BN*DECL), 256, 0, stream>>>(hb2, p3w, p3b, hb3, BN*DECL, 64);
  take_last<<<gblk(BN), 256, 0, stream>>>(hb3, seg_yrul);

  // ---------------- classification heads --------------------------------------
  auto run_head = [&](const HeadP& hp, const float* xin, float* seg) {
    const int M = BN * DECL;
    gemm(xin, hp.w1, hp.b1, hb1, M, 128, DM, 0);
    bnseq_stats<<<DECL, 256, 0, stream>>>(hb1, hm, hv, 128);
    bnseq_relu<<<gblk((size_t)M*128), 256, 0, stream>>>(hb1, hm, hv, hp.g1, hp.be1, 128);
    gemm(hb1, hp.w2, hp.b2, hb2, M, 64, 128, 0);
    bnseq_stats<<<DECL, 256, 0, stream>>>(hb2, hm, hv, 64);
    bnseq_relu<<<gblk((size_t)M*64), 256, 0, stream>>>(hb2, hm, hv, hp.g2, hp.be2, 64);
    gemv1<<<gblk(M), 256, 0, stream>>>(hb2, hp.w3, hp.b3, hb3, M, 64);
    logsoftmax_pos<<<gblk(BN), 256, 0, stream>>>(hb3, seg);
  };
  run_head(clsH, privb,  seg_cls);
  run_head(domH, shareb, seg_dom);

  // share / priv pass-through outputs
  copyf<<<gblk(DEC_N), 256, 0, stream>>>(shareb, seg_share, DEC_N);
  copyf<<<gblk(DEC_N), 256, 0, stream>>>(privb,  seg_priv,  DEC_N);
}
